// MCAM_54992761258365
// MI455X (gfx1250) — matrix operand order repaired and checked by reading, not compiled, not run on hardware
//
#include <hip/hip_runtime.h>
#include <stdint.h>

typedef __attribute__((ext_vector_type(16))) _Float16 v16h;
typedef __attribute__((ext_vector_type(8)))  float    v8f;

union Frag {
  uint32_t u[8];
  uint4    q2[2];
  v16h     h;
};

#define WMMA(A, B, C) \
  __builtin_amdgcn_wmma_f32_16x16x32_f16(false, (A), false, (B), (short)0, (C), false, false)

#define LDSOFF(p) ((uint32_t)(uintptr_t)(p))

__device__ __forceinline__ v8f vzero8() {
  v8f z = {0.f, 0.f, 0.f, 0.f, 0.f, 0.f, 0.f, 0.f};
  return z;
}

// gfx1250 async copy: global -> LDS, 16 bytes per lane, tracked by ASYNCcnt.
__device__ __forceinline__ void async_b128(uint32_t lds_off, const _Float16* g) {
  asm volatile("global_load_async_to_lds_b128 %0, %1, off"
               :: "v"(lds_off), "v"(g) : "memory");
}
// ASYNCcnt completes in order -> waiting to a nonzero threshold implements
// a software pipeline (oldest transfers guaranteed done, newest in flight).
template<int N>
__device__ __forceinline__ void wait_async() {
  asm volatile("s_wait_asynccnt %0" :: "i"(N) : "memory");
}

// ---------------------------------------------------------------------------
// GEMM: Y[M,N] = act(X[M,K] @ W[K,N] + bias), f16 WMMA compute, f32 accum.
// Block tile 128x128, K-tile 32, 8 waves, wave tile 32x64 (2x4 16x16 frags).
// MODE 0: f32 row-major (+RELU)    MODE 1: f16 row-major (Q/K)
// MODE 2: f16 head-transposed VT[(b*8+h)*64+dk][s]  (V)
// ---------------------------------------------------------------------------
#define BM 128
#define BN 128
#define BK 32

template<int MODE, int RELU>
__global__ __launch_bounds__(256) void gemm_kernel(
    const float* __restrict__ X, const float* __restrict__ W,
    const float* __restrict__ bias, float* __restrict__ Yf,
    _Float16* __restrict__ Yh, int N, int K) {
  __shared__ __align__(16) _Float16 As[BM][BK];  // 8 KB
  __shared__ __align__(16) _Float16 Bt[BN][BK];  // 8 KB, W tile transposed

  const int tid  = threadIdx.x;
  const int wave = tid >> 5;
  const int lane = tid & 31;
  const int lh   = lane & 15;
  const int aKb  = (lane < 16) ? 0 : 8;
  const int bKb  = (lane < 16) ? 0 : 8;
  const int mo   = (lane >> 4) << 3;

  const int rowBase = blockIdx.y * BM;
  const int colBase = blockIdx.x * BN;
  const int wm = (wave & 3) * 32;
  const int wn = (wave >> 2) * 64;

  v8f acc[2][4];
  #pragma unroll
  for (int i = 0; i < 2; ++i)
    #pragma unroll
    for (int j = 0; j < 4; ++j) acc[i][j] = vzero8();

  for (int k0 = 0; k0 < K; k0 += BK) {
    {
      const int r = tid >> 3;
      const int c = (tid & 7) * 4;
      #pragma unroll
      for (int i = 0; i < 4; ++i) {
        const float* src = X + (size_t)(rowBase + r + 32 * i) * K + k0 + c;
        float4 v = *(const float4*)src;
        As[r + 32 * i][c + 0] = (_Float16)v.x;
        As[r + 32 * i][c + 1] = (_Float16)v.y;
        As[r + 32 * i][c + 2] = (_Float16)v.z;
        As[r + 32 * i][c + 3] = (_Float16)v.w;
        if (k0 + BK < K) __builtin_prefetch(src + BK, 0, 1);
      }
    }
    {
      const int r = tid >> 5;
      const int c = (tid & 31) * 4;
      #pragma unroll
      for (int i = 0; i < 4; ++i) {
        const float* src = W + (size_t)(k0 + r + 8 * i) * N + colBase + c;
        float4 v = *(const float4*)src;
        Bt[c + 0][r + 8 * i] = (_Float16)v.x;
        Bt[c + 1][r + 8 * i] = (_Float16)v.y;
        Bt[c + 2][r + 8 * i] = (_Float16)v.z;
        Bt[c + 3][r + 8 * i] = (_Float16)v.w;
      }
    }
    __syncthreads();

    Frag a[2], b[4];
    #pragma unroll
    for (int i = 0; i < 2; ++i) {
      const _Float16* p = &As[wm + 16 * i + lh][aKb];
      a[i].q2[0] = *(const uint4*)p;
      a[i].q2[1] = *(const uint4*)(p + 16);
    }
    #pragma unroll
    for (int j = 0; j < 4; ++j) {
      const _Float16* p = &Bt[wn + 16 * j + lh][bKb];
      b[j].q2[0] = *(const uint4*)p;
      b[j].q2[1] = *(const uint4*)(p + 16);
    }
    #pragma unroll
    for (int i = 0; i < 2; ++i)
      #pragma unroll
      for (int j = 0; j < 4; ++j)
        acc[i][j] = WMMA(a[i].h, b[j].h, acc[i][j]);
    __syncthreads();
  }

  #pragma unroll
  for (int i = 0; i < 2; ++i)
    #pragma unroll
    for (int j = 0; j < 4; ++j) {
      const int col = colBase + wn + 16 * j + lh;
      const float bv = bias[col];
      #pragma unroll
      for (int r = 0; r < 8; ++r) {
        const int row = rowBase + wm + 16 * i + mo + r;
        float v = acc[i][j][r] + bv;
        if (MODE == 0) {
          if (RELU) v = fmaxf(v, 0.f);
          Yf[(size_t)row * N + col] = v;
        } else if (MODE == 1) {
          Yh[(size_t)row * N + col] = (_Float16)v;
        } else {
          const int bb = row >> 10, ss = row & 1023;
          const int hh = col >> 6,  dk = col & 63;
          Yh[(((size_t)(bb * 8 + hh) * 64 + dk) << 10) + ss] = (_Float16)v;
        }
      }
    }
}

// ---------------------------------------------------------------------------
// Flash attention: one (b,h), 64 queries/block, 4 waves x 16 q-rows.
// Q,K: f16 [B*S, 512] (head slice at col h*64); VT: f16 [(b*8+h)*64+dk][S].
// Double-buffered GLOBAL_LOAD_ASYNC_TO_LDS_B128 pipeline (4 ops/tile/thread):
// issue tile kt+1, then s_wait_asynccnt 4 -> tile kt complete, kt+1 in flight.
// ---------------------------------------------------------------------------
#define AS_S 1024
#define AS_D 512
#define AS_DK 64
#define QT 64
#define KT 32
#define NT (AS_S / KT)
#define ATT_SCALE 0.125f

__global__ __launch_bounds__(128) void flash_kernel(
    const _Float16* __restrict__ Q, const _Float16* __restrict__ K,
    const _Float16* __restrict__ VT, float* __restrict__ O) {
  __shared__ __align__(16) _Float16 Qs[QT][AS_DK];       // 8 KB
  __shared__ __align__(16) _Float16 Ks[2][KT][AS_DK];    // 8 KB
  __shared__ __align__(16) _Float16 VsT[2][AS_DK][KT];   // 8 KB
  __shared__ __align__(16) _Float16 Ps[4][16][KT];       // 4 KB

  const int tid  = threadIdx.x;
  const int wave = tid >> 5;
  const int lane = tid & 31;
  const int lh   = lane & 15;
  const int aKb  = (lane < 16) ? 0 : 8;
  const int bKb  = (lane < 16) ? 0 : 8;
  const int mo   = (lane >> 4) << 3;

  const int bh = blockIdx.y;
  const int b  = bh >> 3, h = bh & 7;
  const int q0 = blockIdx.x * QT;
  const size_t base  = (size_t)b * AS_S * AS_D + (size_t)h * AS_DK;
  const size_t vbase = (size_t)bh * AS_DK * AS_S;

  auto issue_tile = [&](int kt, int buf) {
    #pragma unroll
    for (int i = 0; i < 2; ++i) {
      const int c = tid + 128 * i;       // 0..255
      const int r = c >> 3, q = c & 7;
      async_b128(LDSOFF(&Ks[buf][r][q * 8]),
                 K + base + (size_t)(kt * KT + r) * AS_D + q * 8);
      const int dk = c >> 2, qq = c & 3;
      async_b128(LDSOFF(&VsT[buf][dk][qq * 8]),
                 VT + vbase + (size_t)dk * AS_S + kt * KT + qq * 8);
    }
  };

  // prologue: async Q tile (4 ops/thread) + first K/V tile (4 ops/thread)
  #pragma unroll
  for (int i = 0; i < 4; ++i) {
    const int c = tid + 128 * i;         // 0..511
    const int r = c >> 3, q = c & 7;
    async_b128(LDSOFF(&Qs[r][q * 8]),
               Q + base + (size_t)(q0 + r) * AS_D + q * 8);
  }
  issue_tile(0, 0);
  wait_async<4>();                       // Q done; tile0 still in flight
  __syncthreads();

  Frag qa[2];
  #pragma unroll
  for (int ki = 0; ki < 2; ++ki) {
    const _Float16* p = &Qs[wave * 16 + lh][ki * 32 + aKb];
    qa[ki].q2[0] = *(const uint4*)p;
    qa[ki].q2[1] = *(const uint4*)(p + 16);
  }

  v8f acc[4];
  float mrow[8], lrow[8];
  #pragma unroll
  for (int j = 0; j < 4; ++j) acc[j] = vzero8();
  #pragma unroll
  for (int r = 0; r < 8; ++r) { mrow[r] = -1e30f; lrow[r] = 0.f; }

  #pragma unroll 2
  for (int kt = 0; kt < NT; ++kt) {
    const int buf = kt & 1;
    if (kt + 1 < NT) {
      issue_tile(kt + 1, buf ^ 1);
      wait_async<4>();                   // tile kt complete, kt+1 in flight
    } else {
      wait_async<0>();
    }
    __syncthreads();

    // scores S = Q K^T
    v8f sf[2];
    #pragma unroll
    for (int j = 0; j < 2; ++j) {
      sf[j] = vzero8();
      Frag kb0, kb1;
      const _Float16* p0 = &Ks[buf][j * 16 + lh][bKb];
      kb0.q2[0] = *(const uint4*)p0;
      kb0.q2[1] = *(const uint4*)(p0 + 16);
      const _Float16* p1 = &Ks[buf][j * 16 + lh][32 + bKb];
      kb1.q2[0] = *(const uint4*)p1;
      kb1.q2[1] = *(const uint4*)(p1 + 16);
      sf[j] = WMMA(qa[0].h, kb0.h, sf[j]);
      sf[j] = WMMA(qa[1].h, kb1.h, sf[j]);
    }

    // online softmax (row = one VGPR across a 16-lane half)
    #pragma unroll
    for (int r = 0; r < 8; ++r) {
      float s0 = sf[0][r] * ATT_SCALE;
      float s1 = sf[1][r] * ATT_SCALE;
      float mx = fmaxf(s0, s1);
      mx = fmaxf(mx, __shfl_xor(mx, 1, 32));
      mx = fmaxf(mx, __shfl_xor(mx, 2, 32));
      mx = fmaxf(mx, __shfl_xor(mx, 4, 32));
      mx = fmaxf(mx, __shfl_xor(mx, 8, 32));
      const float mnew = fmaxf(mrow[r], mx);
      const float corr = __expf(mrow[r] - mnew);
      mrow[r] = mnew;
      const float p0 = __expf(s0 - mnew);
      const float p1 = __expf(s1 - mnew);
      float ps = p0 + p1;
      ps += __shfl_xor(ps, 1, 32);
      ps += __shfl_xor(ps, 2, 32);
      ps += __shfl_xor(ps, 4, 32);
      ps += __shfl_xor(ps, 8, 32);
      lrow[r] = lrow[r] * corr + ps;
      #pragma unroll
      for (int jj = 0; jj < 4; ++jj) acc[jj][r] *= corr;
      Ps[wave][r + mo][lh]      = (_Float16)p0;
      Ps[wave][r + mo][16 + lh] = (_Float16)p1;
    }
    asm volatile("s_wait_dscnt 0" ::: "memory");

    Frag pa;
    {
      const _Float16* p = &Ps[wave][lh][aKb];
      pa.q2[0] = *(const uint4*)p;
      pa.q2[1] = *(const uint4*)(p + 16);
    }
    #pragma unroll
    for (int jj = 0; jj < 4; ++jj) {
      Frag vb;
      const _Float16* p = &VsT[buf][jj * 16 + lh][bKb];
      vb.q2[0] = *(const uint4*)p;
      vb.q2[1] = *(const uint4*)(p + 16);
      acc[jj] = WMMA(pa.h, vb.h, acc[jj]);
    }
    __syncthreads();
  }

  #pragma unroll
  for (int r = 0; r < 8; ++r) {
    const float inv = 1.0f / lrow[r];
    const size_t row = (size_t)(q0 + wave * 16 + r + mo);
    #pragma unroll
    for (int jj = 0; jj < 4; ++jj)
      O[base + row * AS_D + jj * 16 + lh] = acc[jj][r] * inv;
  }
}

// ---------------------------------------------------------------------------
// Dual-softmax cross attention (product of softmaxes, online form):
//   P = pa*pv; Oa += P@Va; Ov += P@Vv; both rescaled by both corrections;
//   ctx = (Oa/(Za*Zv)) * (Ov/(Za*Zv)).  8 async ops/tile/thread pipeline.
// ---------------------------------------------------------------------------
__global__ __launch_bounds__(128) void dual_flash_kernel(
    const _Float16* __restrict__ Qa, const _Float16* __restrict__ Ka,
    const _Float16* __restrict__ VaT, const _Float16* __restrict__ Qv,
    const _Float16* __restrict__ Kv, const _Float16* __restrict__ VvT,
    float* __restrict__ O) {
  __shared__ __align__(16) _Float16 Qas[QT][AS_DK], Qvs[QT][AS_DK];       // 16 KB
  __shared__ __align__(16) _Float16 Kas[2][KT][AS_DK], Kvs[2][KT][AS_DK]; // 16 KB
  __shared__ __align__(16) _Float16 VaS[2][AS_DK][KT], VvS[2][AS_DK][KT]; // 16 KB
  __shared__ __align__(16) _Float16 Ps[4][16][KT];                        //  4 KB

  const int tid  = threadIdx.x;
  const int wave = tid >> 5;
  const int lane = tid & 31;
  const int lh   = lane & 15;
  const int aKb  = (lane < 16) ? 0 : 8;
  const int bKb  = (lane < 16) ? 0 : 8;
  const int mo   = (lane >> 4) << 3;

  const int bh = blockIdx.y;
  const int b  = bh >> 3, h = bh & 7;
  const int q0 = blockIdx.x * QT;
  const size_t base  = (size_t)b * AS_S * AS_D + (size_t)h * AS_DK;
  const size_t vbase = (size_t)bh * AS_DK * AS_S;

  auto issue_tile = [&](int kt, int buf) {
    #pragma unroll
    for (int i = 0; i < 2; ++i) {
      const int c = tid + 128 * i;
      const int r = c >> 3, q = c & 7;
      const size_t gk = base + (size_t)(kt * KT + r) * AS_D + q * 8;
      async_b128(LDSOFF(&Kas[buf][r][q * 8]), Ka + gk);
      async_b128(LDSOFF(&Kvs[buf][r][q * 8]), Kv + gk);
      const int dk = c >> 2, qq = c & 3;
      const size_t gv = vbase + (size_t)dk * AS_S + kt * KT + qq * 8;
      async_b128(LDSOFF(&VaS[buf][dk][qq * 8]), VaT + gv);
      async_b128(LDSOFF(&VvS[buf][dk][qq * 8]), VvT + gv);
    }
  };

  #pragma unroll
  for (int i = 0; i < 4; ++i) {
    const int c = tid + 128 * i;
    const int r = c >> 3, q = c & 7;
    const size_t go = base + (size_t)(q0 + r) * AS_D + q * 8;
    async_b128(LDSOFF(&Qas[r][q * 8]), Qa + go);
    async_b128(LDSOFF(&Qvs[r][q * 8]), Qv + go);
  }
  issue_tile(0, 0);
  wait_async<8>();                       // Q tiles done; tile0 in flight
  __syncthreads();

  Frag qfa[2], qfv[2];
  #pragma unroll
  for (int ki = 0; ki < 2; ++ki) {
    const _Float16* pa = &Qas[wave * 16 + lh][ki * 32 + aKb];
    qfa[ki].q2[0] = *(const uint4*)pa;
    qfa[ki].q2[1] = *(const uint4*)(pa + 16);
    const _Float16* pv = &Qvs[wave * 16 + lh][ki * 32 + aKb];
    qfv[ki].q2[0] = *(const uint4*)pv;
    qfv[ki].q2[1] = *(const uint4*)(pv + 16);
  }

  v8f Oa[4], Ov[4];
  float ma[8], la[8], mv[8], lv[8];
  #pragma unroll
  for (int j = 0; j < 4; ++j) { Oa[j] = vzero8(); Ov[j] = vzero8(); }
  #pragma unroll
  for (int r = 0; r < 8; ++r) {
    ma[r] = -1e30f; la[r] = 0.f; mv[r] = -1e30f; lv[r] = 0.f;
  }

  #pragma unroll 2
  for (int kt = 0; kt < NT; ++kt) {
    const int buf = kt & 1;
    if (kt + 1 < NT) {
      issue_tile(kt + 1, buf ^ 1);
      wait_async<8>();
    } else {
      wait_async<0>();
    }
    __syncthreads();

    v8f sa[2], sv[2];
    #pragma unroll
    for (int j = 0; j < 2; ++j) {
      sa[j] = vzero8(); sv[j] = vzero8();
      Frag k0, k1;
      const _Float16* p;
      p = &Kas[buf][j * 16 + lh][bKb];
      k0.q2[0] = *(const uint4*)p; k0.q2[1] = *(const uint4*)(p + 16);
      p = &Kas[buf][j * 16 + lh][32 + bKb];
      k1.q2[0] = *(const uint4*)p; k1.q2[1] = *(const uint4*)(p + 16);
      sa[j] = WMMA(qfa[0].h, k0.h, sa[j]);
      sa[j] = WMMA(qfa[1].h, k1.h, sa[j]);
      p = &Kvs[buf][j * 16 + lh][bKb];
      k0.q2[0] = *(const uint4*)p; k0.q2[1] = *(const uint4*)(p + 16);
      p = &Kvs[buf][j * 16 + lh][32 + bKb];
      k1.q2[0] = *(const uint4*)p; k1.q2[1] = *(const uint4*)(p + 16);
      sv[j] = WMMA(qfv[0].h, k0.h, sv[j]);
      sv[j] = WMMA(qfv[1].h, k1.h, sv[j]);
    }

    #pragma unroll
    for (int r = 0; r < 8; ++r) {
      float a0 = sa[0][r] * ATT_SCALE, a1 = sa[1][r] * ATT_SCALE;
      float mxa = fmaxf(a0, a1);
      mxa = fmaxf(mxa, __shfl_xor(mxa, 1, 32));
      mxa = fmaxf(mxa, __shfl_xor(mxa, 2, 32));
      mxa = fmaxf(mxa, __shfl_xor(mxa, 4, 32));
      mxa = fmaxf(mxa, __shfl_xor(mxa, 8, 32));
      const float mna = fmaxf(ma[r], mxa);
      const float ca  = __expf(ma[r] - mna);
      ma[r] = mna;
      const float pa0 = __expf(a0 - mna), pa1 = __expf(a1 - mna);
      float psa = pa0 + pa1;
      psa += __shfl_xor(psa, 1, 32);
      psa += __shfl_xor(psa, 2, 32);
      psa += __shfl_xor(psa, 4, 32);
      psa += __shfl_xor(psa, 8, 32);
      la[r] = la[r] * ca + psa;

      float v0 = sv[0][r] * ATT_SCALE, v1 = sv[1][r] * ATT_SCALE;
      float mxv = fmaxf(v0, v1);
      mxv = fmaxf(mxv, __shfl_xor(mxv, 1, 32));
      mxv = fmaxf(mxv, __shfl_xor(mxv, 2, 32));
      mxv = fmaxf(mxv, __shfl_xor(mxv, 4, 32));
      mxv = fmaxf(mxv, __shfl_xor(mxv, 8, 32));
      const float mnv = fmaxf(mv[r], mxv);
      const float cv  = __expf(mv[r] - mnv);
      mv[r] = mnv;
      const float pv0 = __expf(v0 - mnv), pv1 = __expf(v1 - mnv);
      float psv = pv0 + pv1;
      psv += __shfl_xor(psv, 1, 32);
      psv += __shfl_xor(psv, 2, 32);
      psv += __shfl_xor(psv, 4, 32);
      psv += __shfl_xor(psv, 8, 32);
      lv[r] = lv[r] * cv + psv;

      const float c2 = ca * cv;
      #pragma unroll
      for (int jj = 0; jj < 4; ++jj) { Oa[jj][r] *= c2; Ov[jj][r] *= c2; }
      Ps[wave][r + mo][lh]      = (_Float16)(pa0 * pv0);
      Ps[wave][r + mo][16 + lh] = (_Float16)(pa1 * pv1);
    }
    asm volatile("s_wait_dscnt 0" ::: "memory");

    Frag pf;
    {
      const _Float16* p = &Ps[wave][lh][aKb];
      pf.q2[0] = *(const uint4*)p;
      pf.q2[1] = *(const uint4*)(p + 16);
    }
    #pragma unroll
    for (int jj = 0; jj < 4; ++jj) {
      Frag vb;
      const _Float16* p = &VaS[buf][jj * 16 + lh][bKb];
      vb.q2[0] = *(const uint4*)p; vb.q2[1] = *(const uint4*)(p + 16);
      Oa[jj] = WMMA(pf.h, vb.h, Oa[jj]);
      p = &VvS[buf][jj * 16 + lh][bKb];
      vb.q2[0] = *(const uint4*)p; vb.q2[1] = *(const uint4*)(p + 16);
      Ov[jj] = WMMA(pf.h, vb.h, Ov[jj]);
    }
    __syncthreads();
  }

  #pragma unroll
  for (int r = 0; r < 8; ++r) {
    const float inv = 1.0f / (la[r] * lv[r]);   // 1/(Za*Zv)
    const size_t row = (size_t)(q0 + wave * 16 + r + mo);
    #pragma unroll
    for (int jj = 0; jj < 4; ++jj)
      O[base + row * AS_D + jj * 16 + lh] =
          (Oa[jj][r] * inv) * (Ov[jj][r] * inv);
  }
}

// ---------------------------------------------------------------------------
// LayerNorm over rows of 512: y = LN(x1 [+x2] [+x3] [+x4]) * g + b
// ---------------------------------------------------------------------------
__global__ __launch_bounds__(256) void ln_kernel(
    const float* __restrict__ x1, const float* __restrict__ x2,
    const float* __restrict__ x3, const float* __restrict__ x4,
    const float* __restrict__ g, const float* __restrict__ bta,
    float* __restrict__ y) {
  const int row  = blockIdx.x * 8 + (threadIdx.x >> 5);
  const int lane = threadIdx.x & 31;
  const size_t off = (size_t)row * 512;
  float v[16];
  float s = 0.f;
  #pragma unroll
  for (int i = 0; i < 16; ++i) {
    const int c = i * 32 + lane;
    float t = x1[off + c];
    if (x2) t += x2[off + c];
    if (x3) t += x3[off + c];
    if (x4) t += x4[off + c];
    v[i] = t;
    s += t;
  }
  #pragma unroll
  for (int o = 1; o < 32; o <<= 1) s += __shfl_xor(s, o, 32);
  const float mean = s * (1.0f / 512.0f);
  float sq = 0.f;
  #pragma unroll
  for (int i = 0; i < 16; ++i) {
    const float d = v[i] - mean;
    sq += d * d;
  }
  #pragma unroll
  for (int o = 1; o < 32; o <<= 1) sq += __shfl_xor(sq, o, 32);
  const float rstd = rsqrtf(sq * (1.0f / 512.0f) + 1e-5f);
  #pragma unroll
  for (int i = 0; i < 16; ++i) {
    const int c = i * 32 + lane;
    y[off + c] = (v[i] - mean) * rstd * g[c] + bta[c];
  }
}

// ---------------------------------------------------------------------------
// Orchestration
// ---------------------------------------------------------------------------
extern "C" void kernel_launch(void* const* d_in, const int* in_sizes, int n_in,
                              void* d_out, int out_size, void* d_ws, size_t ws_size,
                              hipStream_t stream) {
  (void)in_sizes; (void)n_in; (void)out_size; (void)ws_size;

  auto in = [&](int i) { return (const float*)d_in[i]; };
  const float* audio = in(0);
  const float* video = in(1);

  float* ws = (float*)d_ws;
  const size_t SLOT = (size_t)4096 * 512;
  auto slot = [&](int i) { return ws + (size_t)i * SLOT; };          // f32, 6 slots
  _Float16* hws = (_Float16*)(ws + 6 * SLOT);
  auto hslot = [&](int i) { return hws + (size_t)i * SLOT; };        // f16, 6 slots

  const dim3 gg(512 / BN, 4096 / BM);
  auto gemm_f = [&](const float* X, const float* W, const float* B, float* Y) {
    gemm_kernel<0, 0><<<gg, 256, 0, stream>>>(X, W, B, Y, nullptr, 512, 512);
  };
  auto gemm_relu = [&](const float* X, const float* W, const float* B, float* Y) {
    gemm_kernel<0, 1><<<gg, 256, 0, stream>>>(X, W, B, Y, nullptr, 512, 512);
  };
  auto gemm_h = [&](const float* X, const float* W, const float* B, _Float16* Y) {
    gemm_kernel<1, 0><<<gg, 256, 0, stream>>>(X, W, B, nullptr, Y, 512, 512);
  };
  auto gemm_vt = [&](const float* X, const float* W, const float* B, _Float16* Y) {
    gemm_kernel<2, 0><<<gg, 256, 0, stream>>>(X, W, B, nullptr, Y, 512, 512);
  };
  auto ln = [&](const float* a, const float* b, const float* c, const float* d,
                const float* g2, const float* b2, float* y) {
    ln_kernel<<<4096 / 8, 256, 0, stream>>>(a, b, c, d, g2, b2, y);
  };

  // Post-LN encoder layer. Weight base wb (dict order):
  //   +0 qw +1 kw +2 vw +3 ow +4 qb +5 kb +6 vb +7 ob
  //   +8 w1 +9 c1 +10 w2 +11 c2 +12 g1 +13 b1 +14 g2 +15 b2
  auto encoder = [&](const float* x, int wb, float* out) {
    gemm_h(x, in(wb + 0), in(wb + 4), hslot(0));   // Q   (f16)
    gemm_h(x, in(wb + 1), in(wb + 5), hslot(1));   // K   (f16)
    gemm_vt(x, in(wb + 2), in(wb + 6), hslot(2));  // V^T (f16, per-head)
    flash_kernel<<<dim3(16, 32), 128, 0, stream>>>(hslot(0), hslot(1), hslot(2), slot(0));
    gemm_f(slot(0), in(wb + 3), in(wb + 7), slot(1));                     // ctx @ ow + ob
    ln(slot(1), x, nullptr, nullptr, in(wb + 12), in(wb + 13), slot(2));  // h
    gemm_relu(slot(2), in(wb + 8), in(wb + 9), slot(1));                  // relu(h@w1+c1)
    gemm_f(slot(1), in(wb + 10), in(wb + 11), slot(3));                   // @w2+c2
    ln(slot(3), slot(2), nullptr, nullptr, in(wb + 14), in(wb + 15), out);
  };

  encoder(audio, 18, slot(4));   // audio_self  (asa_* at idx 18..33)
  encoder(video, 34, slot(5));   // video_self  (vsa_* at idx 34..49)

  // Cross projections (aq/ak/av/vq/vk/vv at idx 2..13)
  gemm_h (audio, in(2),  in(3),  hslot(0));  // Qa
  gemm_h (audio, in(4),  in(5),  hslot(1));  // Ka
  gemm_vt(audio, in(6),  in(7),  hslot(2));  // Va^T
  gemm_h (video, in(8),  in(9),  hslot(3));  // Qv
  gemm_h (video, in(10), in(11), hslot(4));  // Kv
  gemm_vt(video, in(12), in(13), hslot(5));  // Vv^T

  dual_flash_kernel<<<dim3(16, 32), 128, 0, stream>>>(
      hslot(0), hslot(1), hslot(2), hslot(3), hslot(4), hslot(5), slot(0));

  gemm_f(slot(0), in(14), in(15), slot(1));  // context = ctx @ out_w + out_b

  // final: LN(audio + context + audio_self + video_self)
  ln(audio, slot(1), slot(4), slot(5), in(16), in(17), (float*)d_out);
}
